// Attention_27255862460439
// MI455X (gfx1250) — compile-verified, matrix operand order checked by reading, not run on hardware
//
#include <hip/hip_runtime.h>
#include <hip/hip_bf16.h>

typedef __bf16 bf16_t;
typedef bf16_t v16bf __attribute__((ext_vector_type(16)));
typedef float  v8f   __attribute__((ext_vector_type(8)));
typedef int    v4i_t __attribute__((vector_size(16)));   // int4, matches builtin param

#define DEVINL static __device__ __forceinline__

#define ATT_SCALE 0.125f   // (1024/16)^-0.5

#define HAS_ASYNC_LDS __has_builtin(__builtin_amdgcn_global_load_async_to_lds_b128)

DEVINL v8f vzero8() {
  v8f z;
#pragma unroll
  for (int i = 0; i < 8; ++i) z[i] = 0.0f;
  return z;
}

DEVINL v8f wmma_bf16(v16bf a, v16bf b, v8f c) {
  // D = A(16x32) * B(32x16) + C(16x16), fp32 accumulate
  return __builtin_amdgcn_wmma_f32_16x16x32_bf16(
      /*neg_a=*/false, a, /*neg_b=*/false, b,
      /*c_mod=*/(short)0, c, /*reuse_a=*/false, /*reuse_b=*/false);
}

// Copy 8 bf16 (16 bytes) global -> LDS. On CDNA5 use the async path
// (GLOBAL_LOAD_ASYNC_TO_LDS_B128, tracked by ASYNCcnt); else fall back to
// a VGPR round trip.
DEVINL void cp16b_to_lds(const bf16_t* g, bf16_t* l) {
#if HAS_ASYNC_LDS
  __builtin_amdgcn_global_load_async_to_lds_b128(
      (v4i_t*)(bf16_t*)g,
      (__attribute__((address_space(3))) v4i_t*)l,
      /*imm offset=*/0, /*cpol=*/0);
#else
#pragma unroll
  for (int e = 0; e < 8; ++e) l[e] = g[e];
#endif
}

// Drain outstanding async LDS transfers, then workgroup barrier.
DEVINL void stage_commit_barrier() {
#if HAS_ASYNC_LDS && __has_builtin(__builtin_amdgcn_s_wait_asynccnt)
  __builtin_amdgcn_s_wait_asynccnt(0);
#endif
  __syncthreads();
}

// 16-bit A fragment 16x32 (MxK), ISA 7.12.2:
//   lanes 0-15: m=lane, VGPR0..3 -> K 0..7, VGPR4..7 -> K 16..23
//   lanes16-31: m=lane-16, VGPR0..3 -> K 8..15, VGPR4..7 -> K 24..31
DEVINL v16bf load_frag_a(const bf16_t* __restrict__ smem, int ld,
                         int row0, int k0, int lane) {
  const int half = lane >> 4;
  const int m = lane & 15;
  v16bf a;
#pragma unroll
  for (int e = 0; e < 16; ++e) {
    int k = ((e & 8) << 1) + half * 8 + (e & 7);
    a[e] = smem[(row0 + m) * ld + k0 + k];
  }
  return a;
}

// 16-bit B fragment 32x16 (KxN): mirror of A with n = lane&15
DEVINL v16bf load_frag_b(const bf16_t* __restrict__ smem, int ld,
                         int k0, int col0, int lane) {
  const int half = lane >> 4;
  const int n = lane & 15;
  v16bf b;
#pragma unroll
  for (int e = 0; e < 16; ++e) {
    int k = ((e & 8) << 1) + half * 8 + (e & 7);
    b[e] = smem[(k0 + k) * ld + col0 + n];
  }
  return b;
}

// ---------------------------------------------------------------- converts
__global__ void f32_to_bf16_kernel(const float* __restrict__ src,
                                   bf16_t* __restrict__ dst, int n) {
  int i = blockIdx.x * blockDim.x + threadIdx.x;
  int stride = gridDim.x * blockDim.x;
  for (; i < n; i += stride) dst[i] = (bf16_t)src[i];
}

// ------------------------------------------------- LoRA A: x@w_a (rank 4)
__global__ __launch_bounds__(256) void lora_a_kernel(
    const float* __restrict__ x,      // [8192][1024]
    const float* __restrict__ waq,    // [1024][4]
    const float* __restrict__ wav,    // [1024][4]
    float* __restrict__ aq,           // [8192][4]
    float* __restrict__ av) {         // [8192][4]
  int m = blockIdx.x * blockDim.x + threadIdx.x;
  if (m >= 8192) return;
  float sq[4] = {0.f, 0.f, 0.f, 0.f};
  float sv[4] = {0.f, 0.f, 0.f, 0.f};
  const float* xr = x + (size_t)m * 1024;
  for (int c = 0; c < 1024; ++c) {
    float xv = xr[c];
#pragma unroll
    for (int r = 0; r < 4; ++r) {
      sq[r] += xv * waq[c * 4 + r];
      sv[r] += xv * wav[c * 4 + r];
    }
  }
#pragma unroll
  for (int r = 0; r < 4; ++r) {
    aq[m * 4 + r] = sq[r];
    av[m * 4 + r] = sv[r];
  }
}

// ----------------------------------------- qkv = x@Wqkv (+LoRA), head split
// Block tile 128x128, 8 waves each 32x64 (2x4 WMMA tiles), K slab = 32.
__global__ __launch_bounds__(256) void qkv_gemm_kernel(
    const bf16_t* __restrict__ A,    // xb [8192][1024]
    const bf16_t* __restrict__ Bw,   // wqkv [1024][3072]
    const float*  __restrict__ aq,   // [8192][4]
    const float*  __restrict__ av,   // [8192][4]
    const float*  __restrict__ wbq,  // [4][1024] fp32
    const float*  __restrict__ wbv,  // [4][1024] fp32
    bf16_t* __restrict__ qout,       // [64][2048][64]  (bh, n, d)
    bf16_t* __restrict__ ktout,      // [64][64][2048]  (bh, d, n)
    bf16_t* __restrict__ vout) {     // [64][2048][64]
  __shared__ bf16_t As[128][40];
  __shared__ bf16_t Bs[32][136];
  const int tid = threadIdx.x;
  const int lane = tid & 31;
  const int wave = tid >> 5;
  const int m_blk = blockIdx.x * 128;
  const int n_blk = blockIdx.y * 128;
  const int wm = wave >> 1;  // 0..3
  const int wn = wave & 1;   // 0..1

  v8f acc[2][4];
#pragma unroll
  for (int mi = 0; mi < 2; ++mi)
#pragma unroll
    for (int ni = 0; ni < 4; ++ni) acc[mi][ni] = vzero8();

  for (int k0 = 0; k0 < 1024; k0 += 32) {
    {  // stage A slab 128x32 (async path -> LDS)
      int r = tid >> 1, c = (tid & 1) * 16;
      const bf16_t* src = A + (size_t)(m_blk + r) * 1024 + k0 + c;
      __builtin_prefetch(src + 32);
      cp16b_to_lds(src, &As[r][c]);
      cp16b_to_lds(src + 8, &As[r][c + 8]);
    }
    {  // stage B slab 32x128
      int r = tid >> 3, c = (tid & 7) * 16;
      const bf16_t* src = Bw + (size_t)(k0 + r) * 3072 + n_blk + c;
      __builtin_prefetch(src + 32 * 3072);
      cp16b_to_lds(src, &Bs[r][c]);
      cp16b_to_lds(src + 8, &Bs[r][c + 8]);
    }
    stage_commit_barrier();
    v16bf bf[4];
#pragma unroll
    for (int ni = 0; ni < 4; ++ni)
      bf[ni] = load_frag_b(&Bs[0][0], 136, 0, wn * 64 + ni * 16, lane);
#pragma unroll
    for (int mi = 0; mi < 2; ++mi) {
      v16bf af = load_frag_a(&As[0][0], 40, wm * 32 + mi * 16, 0, lane);
#pragma unroll
      for (int ni = 0; ni < 4; ++ni) acc[mi][ni] = wmma_bf16(af, bf[ni], acc[mi][ni]);
    }
    __syncthreads();
  }

  // epilogue: LoRA add + per-head scatter
#pragma unroll
  for (int mi = 0; mi < 2; ++mi)
#pragma unroll
    for (int ni = 0; ni < 4; ++ni)
#pragma unroll
      for (int r = 0; r < 8; ++r) {
        int m = m_blk + wm * 32 + mi * 16 + r + 8 * (lane >> 4);
        int n = n_blk + wn * 64 + ni * 16 + (lane & 15);
        float val = acc[mi][ni][r];
        int sec = n >> 10;       // 0=q 1=k 2=v
        int c = n & 1023;
        int h = c >> 6, d = c & 63;
        int b = m >> 11, nn = m & 2047;
        int bh = b * 16 + h;
        if (sec == 0) {
          float lora = 0.f;
#pragma unroll
          for (int rr = 0; rr < 4; ++rr) lora += aq[m * 4 + rr] * wbq[rr * 1024 + c];
          qout[((size_t)bh * 2048 + nn) * 64 + d] = (bf16_t)(val + lora);
        } else if (sec == 1) {
          ktout[((size_t)bh * 64 + d) * 2048 + nn] = (bf16_t)val;
        } else {
          float lora = 0.f;
#pragma unroll
          for (int rr = 0; rr < 4; ++rr) lora += av[m * 4 + rr] * wbv[rr * 1024 + c];
          vout[((size_t)bh * 2048 + nn) * 64 + d] = (bf16_t)(val + lora);
        }
      }
}

// ------------------------------------- flash attention, one (bh, 128 rows)
__global__ __launch_bounds__(256) void attention_kernel(
    const bf16_t* __restrict__ Q,   // [64][2048][64]
    const bf16_t* __restrict__ KT,  // [64][64][2048]
    const bf16_t* __restrict__ V,   // [64][2048][64]
    bf16_t* __restrict__ O) {       // [64][2048][64]
  __shared__ bf16_t Ks[64][72];        // [d][key]   64-key slab
  __shared__ bf16_t Vs[64][72];        // [key][d]
  __shared__ bf16_t Ps[8][16][72];     // per-wave P tile (16 rows x 64 keys)
  const int tid = threadIdx.x;
  const int lane = tid & 31;
  const int wave = tid >> 5;
  const int bh = blockIdx.y;
  const int q0 = blockIdx.x * 128;
  const bf16_t* qp = Q + (size_t)bh * 2048 * 64;
  const bf16_t* kp = KT + (size_t)bh * 64 * 2048;
  const bf16_t* vp = V + (size_t)bh * 2048 * 64;

  // per-wave Q A-fragments (16 rows, K=64 -> 2 fragments), loaded once
  v16bf qf[2];
  {
    const int half = lane >> 4;
    const bf16_t* qrow = qp + (size_t)(q0 + wave * 16 + (lane & 15)) * 64;
#pragma unroll
    for (int s = 0; s < 2; ++s)
#pragma unroll
      for (int e = 0; e < 16; ++e) {
        int k = s * 32 + ((e & 8) << 1) + half * 8 + (e & 7);
        qf[s][e] = qrow[k];
      }
  }

  v8f acc[4];
#pragma unroll
  for (int dj = 0; dj < 4; ++dj) acc[dj] = vzero8();
  float rmax[8], rsum[8];
#pragma unroll
  for (int r = 0; r < 8; ++r) { rmax[r] = -3.0e38f; rsum[r] = 0.f; }

  for (int kb = 0; kb < 2048; kb += 64) {
    {  // stage K^T slab [d=64][key=64] (async -> LDS)
      int r = tid >> 2, c = (tid & 3) * 16;
      const bf16_t* src = kp + (size_t)r * 2048 + kb + c;
      cp16b_to_lds(src, &Ks[r][c]);
      cp16b_to_lds(src + 8, &Ks[r][c + 8]);
    }
    {  // stage V slab [key=64][d=64]
      int r = tid >> 2, c = (tid & 3) * 16;
      const bf16_t* src = vp + (size_t)(kb + r) * 64 + c;
      cp16b_to_lds(src, &Vs[r][c]);
      cp16b_to_lds(src + 8, &Vs[r][c + 8]);
    }
    stage_commit_barrier();

    // S = q @ k^T : 4 key-col tiles x 2 K-slabs
    v8f s[4];
#pragma unroll
    for (int j = 0; j < 4; ++j) {
      s[j] = vzero8();
#pragma unroll
      for (int ks = 0; ks < 2; ++ks) {
        v16bf bfrag = load_frag_b(&Ks[0][0], 72, ks * 32, j * 16, lane);
        s[j] = wmma_bf16(qf[ks], bfrag, s[j]);
      }
    }

    // online softmax: row stats live per (r, lane-half); reduce over 16 lanes
#pragma unroll
    for (int r = 0; r < 8; ++r) {
      float mx = rmax[r];
#pragma unroll
      for (int j = 0; j < 4; ++j) mx = fmaxf(mx, s[j][r] * ATT_SCALE);
#pragma unroll
      for (int off = 1; off < 16; off <<= 1) mx = fmaxf(mx, __shfl_xor(mx, off, 32));
      float corr = __expf(rmax[r] - mx);
      rsum[r] *= corr;
#pragma unroll
      for (int dj = 0; dj < 4; ++dj) acc[dj][r] = acc[dj][r] * corr;
      rmax[r] = mx;
    }
#pragma unroll
    for (int r = 0; r < 8; ++r) {
      float psum = 0.f;
      int prow = r + 8 * (lane >> 4);
#pragma unroll
      for (int j = 0; j < 4; ++j) {
        float p = __expf(s[j][r] * ATT_SCALE - rmax[r]);
        psum += p;
        Ps[wave][prow][j * 16 + (lane & 15)] = (bf16_t)p;
      }
#pragma unroll
      for (int off = 1; off < 16; off <<= 1) psum += __shfl_xor(psum, off, 32);
      rsum[r] += psum;
    }

    // O += P @ V  (per-wave Ps region; DS ops are in-order within a wave)
#pragma unroll
    for (int ks = 0; ks < 2; ++ks) {
      v16bf pfrag = load_frag_a(&Ps[wave][0][0], 72, 0, ks * 32, lane);
#pragma unroll
      for (int dj = 0; dj < 4; ++dj) {
        v16bf vfrag = load_frag_b(&Vs[0][0], 72, ks * 32, dj * 16, lane);
        acc[dj] = wmma_bf16(pfrag, vfrag, acc[dj]);
      }
    }
    __syncthreads();
  }

  // finalize: divide by row sum, store bf16 [bh][n][d]
  bf16_t* op = O + (size_t)bh * 2048 * 64;
#pragma unroll
  for (int dj = 0; dj < 4; ++dj)
#pragma unroll
    for (int r = 0; r < 8; ++r) {
      int m = q0 + wave * 16 + r + 8 * (lane >> 4);
      int d = dj * 16 + (lane & 15);
      op[(size_t)m * 64 + d] = (bf16_t)(acc[dj][r] / rsum[r]);
    }
}

// -------------------------------------------- out = attn_out @ Wproj + b
__global__ __launch_bounds__(256) void proj_gemm_kernel(
    const bf16_t* __restrict__ Ain,  // [64][2048][64] (bh, n, d)
    const bf16_t* __restrict__ Bw,   // wproj [1024][1024]
    const float* __restrict__ bias,  // [1024]
    float* __restrict__ out) {       // [8192][1024]
  __shared__ bf16_t As[128][40];
  __shared__ bf16_t Bs[32][136];
  const int tid = threadIdx.x;
  const int lane = tid & 31;
  const int wave = tid >> 5;
  const int m_blk = blockIdx.x * 128;
  const int n_blk = blockIdx.y * 128;
  const int wm = wave >> 1;
  const int wn = wave & 1;

  v8f acc[2][4];
#pragma unroll
  for (int mi = 0; mi < 2; ++mi)
#pragma unroll
    for (int ni = 0; ni < 4; ++ni) acc[mi][ni] = vzero8();

  for (int k0 = 0; k0 < 1024; k0 += 32) {
    {  // stage A: gather from per-head layout (16 contiguous d within a head)
      int r = tid >> 1;
      int c0 = k0 + (tid & 1) * 16;
      int m = m_blk + r;
      int b = m >> 11, nn = m & 2047;
      int h = c0 >> 6, d0 = c0 & 63;
      const bf16_t* src = Ain + ((size_t)(b * 16 + h) * 2048 + nn) * 64 + d0;
      cp16b_to_lds(src, &As[r][(tid & 1) * 16]);
      cp16b_to_lds(src + 8, &As[r][(tid & 1) * 16 + 8]);
    }
    {  // stage B slab 32x128
      int r = tid >> 3, c = (tid & 7) * 16;
      const bf16_t* src = Bw + (size_t)(k0 + r) * 1024 + n_blk + c;
      __builtin_prefetch(src + 32 * 1024);
      cp16b_to_lds(src, &Bs[r][c]);
      cp16b_to_lds(src + 8, &Bs[r][c + 8]);
    }
    stage_commit_barrier();
    v16bf bf[4];
#pragma unroll
    for (int ni = 0; ni < 4; ++ni)
      bf[ni] = load_frag_b(&Bs[0][0], 136, 0, wn * 64 + ni * 16, lane);
#pragma unroll
    for (int mi = 0; mi < 2; ++mi) {
      v16bf af = load_frag_a(&As[0][0], 40, wm * 32 + mi * 16, 0, lane);
#pragma unroll
      for (int ni = 0; ni < 4; ++ni) acc[mi][ni] = wmma_bf16(af, bf[ni], acc[mi][ni]);
    }
    __syncthreads();
  }

#pragma unroll
  for (int mi = 0; mi < 2; ++mi)
#pragma unroll
    for (int ni = 0; ni < 4; ++ni)
#pragma unroll
      for (int r = 0; r < 8; ++r) {
        int m = m_blk + wm * 32 + mi * 16 + r + 8 * (lane >> 4);
        int n = n_blk + wn * 64 + ni * 16 + (lane & 15);
        out[(size_t)m * 1024 + n] = acc[mi][ni][r] + bias[n];
      }
}

// ---------------------------------------------------------------- launch
extern "C" void kernel_launch(void* const* d_in, const int* in_sizes, int n_in,
                              void* d_out, int out_size, void* d_ws, size_t ws_size,
                              hipStream_t stream) {
  const float* x      = (const float*)d_in[0];
  const float* w_qkv  = (const float*)d_in[1];
  const float* w_a_q  = (const float*)d_in[2];
  const float* w_b_q  = (const float*)d_in[3];
  const float* w_a_v  = (const float*)d_in[4];
  const float* w_b_v  = (const float*)d_in[5];
  const float* w_proj = (const float*)d_in[6];
  const float* b_proj = (const float*)d_in[7];
  float* out = (float*)d_out;
  (void)in_sizes; (void)n_in; (void)out_size; (void)ws_size;

  char* ws = (char*)d_ws;
  size_t off = 0;
  auto carve = [&](size_t bytes) -> void* {
    void* p = (void*)(ws + off);
    off += (bytes + 255) & ~(size_t)255;
    return p;
  };
  bf16_t* xb     = (bf16_t*)carve(8192ull * 1024 * 2);
  bf16_t* wqkvb  = (bf16_t*)carve(1024ull * 3072 * 2);
  bf16_t* wprojb = (bf16_t*)carve(1024ull * 1024 * 2);
  float*  aq     = (float*)carve(8192ull * 4 * 4);
  float*  av     = (float*)carve(8192ull * 4 * 4);
  bf16_t* qb     = (bf16_t*)carve(8388608ull * 2);
  bf16_t* ktb    = (bf16_t*)carve(8388608ull * 2);
  bf16_t* vb     = (bf16_t*)carve(8388608ull * 2);
  bf16_t* ob     = (bf16_t*)carve(8388608ull * 2);

  f32_to_bf16_kernel<<<1024, 256, 0, stream>>>(x, xb, 8192 * 1024);
  f32_to_bf16_kernel<<<1024, 256, 0, stream>>>(w_qkv, wqkvb, 1024 * 3072);
  f32_to_bf16_kernel<<<256, 256, 0, stream>>>(w_proj, wprojb, 1024 * 1024);
  lora_a_kernel<<<32, 256, 0, stream>>>(x, w_a_q, w_a_v, aq, av);
  qkv_gemm_kernel<<<dim3(64, 24), 256, 0, stream>>>(xb, wqkvb, aq, av,
                                                    w_b_q, w_b_v, qb, ktb, vb);
  attention_kernel<<<dim3(16, 64), 256, 0, stream>>>(qb, ktb, vb, ob);
  proj_gemm_kernel<<<dim3(64, 8), 256, 0, stream>>>(ob, wprojb, b_proj, out);
}
